// PGNN_plus_64003602645179
// MI455X (gfx1250) — matrix operand order repaired and checked by reading, not compiled, not run on hardware
//
#include <hip/hip_runtime.h>

typedef __attribute__((ext_vector_type(16))) _Float16 v16h;
typedef __attribute__((ext_vector_type(8)))  float    v8f;

// =====================================================================
// WMMA GEMM:  C[N,MD] = A1[N,KD] @ B1[KD,MD]  (+ A2[N,KD] @ B2[KD,MD]) + bias
// One wave per 16x16 C tile; block covers a 16-row stripe across all MD cols.
// f16 operands (converted in LDS), f32 accumulation via v_wmma_f32_16x16x32_f16.
// NOTE: requires N % 16 == 0 (true here: N = 20000), so row bounds are
// statically satisfied and the load/store paths are branch-free.
// =====================================================================
template<int KD, int MD, bool DUAL>
__global__ __launch_bounds__(32*(MD/16))
void gemm_wmma_kernel(const float* __restrict__ A1, const float* __restrict__ B1,
                      const float* __restrict__ A2, const float* __restrict__ B2,
                      const float* __restrict__ bias, float* __restrict__ C)
{
    constexpr int NW   = MD / 16;          // waves per block
    constexpr int KT   = DUAL ? 2*KD : KD; // staged K extent
    constexpr int LDA  = KT + 8;           // pad (mult of 8 halves -> 16B aligned rows)
    constexpr int LDB  = MD + 8;
    constexpr int NTHR = 32*NW;

    __shared__ _Float16 sA[16*LDA];
    __shared__ _Float16 sB[KT*LDB];

    const int row0 = blockIdx.x * 16;
    const int tid  = threadIdx.x;

    // Stage A tile (16 x KT), converting fp32 -> f16 (branch-free, full tiles)
    for (int idx = tid; idx < 16*KD; idx += NTHR) {
        int m = idx / KD, k = idx % KD;
        int gr = row0 + m;
        sA[m*LDA + k] = (_Float16)A1[(size_t)gr*KD + k];
        if constexpr (DUAL) sA[m*LDA + KD + k] = (_Float16)A2[(size_t)gr*KD + k];
    }
    // Stage weights (KT x MD)
    for (int idx = tid; idx < KD*MD; idx += NTHR) {
        int k = idx / MD, n = idx % MD;
        sB[k*LDB + n] = (_Float16)B1[k*MD + n];
        if constexpr (DUAL) sB[(KD + k)*LDB + n] = (_Float16)B2[k*MD + n];
    }
    __syncthreads();

    const int wave = tid >> 5, lane = tid & 31;
    const int n0   = wave * 16;
    const int nloc = lane & 15;     // C column within tile
    const int hsel = lane >> 4;

    v8f c;
    float bv = bias ? bias[n0 + nloc] : 0.0f;
    #pragma unroll
    for (int r = 0; r < 8; ++r) c[r] = bv;

    #pragma unroll
    for (int kb = 0; kb < KT; kb += 32) {
        v16h a, b;
        // A fragment: lane holds row (lane&15); K order per ISA 16-bit A layout
        #pragma unroll
        for (int v = 0; v < 8; ++v) {
            int k = kb + ((v < 4) ? (8*hsel + 2*v) : (16 + 8*hsel + 2*(v-4)));
            a[2*v]   = sA[nloc*LDA + k];
            a[2*v+1] = sA[nloc*LDA + k + 1];
        }
        // B fragment: lane holds column n0+(lane&15); K order matches A (ISA 16-bit layout)
        #pragma unroll
        for (int j = 0; j < 16; ++j) b[j] = sB[(kb + ((j < 8) ? (8*hsel + j) : (16 + 8*hsel + (j-8))))*LDB + n0 + nloc];
        c = __builtin_amdgcn_wmma_f32_16x16x32_f16(false, a, false, b,
                                                   (short)0, c, false, false);
    }

    // Straight-line coalesced epilogue (no bounds checks: full tiles only)
    #pragma unroll
    for (int r = 0; r < 8; ++r) {
        int gr = row0 + r + 8*hsel;
        C[(size_t)gr*MD + n0 + nloc] = c[r];
    }
}

// =====================================================================
// Trainable distance nonlinearity: d = relu(t*w1 + b1) @ w2 + b2, per (i,k)
// =====================================================================
__global__ void dist_nl_kernel(const float* __restrict__ dmax, const float* __restrict__ w1,
                               const float* __restrict__ b1, const float* __restrict__ w2,
                               const float* __restrict__ b2, float* __restrict__ d,
                               int total, int DHH)
{
    int i = blockIdx.x*blockDim.x + threadIdx.x;
    if (i >= total) return;
    float t = dmax[i];
    float acc = b2[0];
    for (int f = 0; f < DHH; ++f)
        acc += fmaxf(t*w1[f] + b1[f], 0.0f) * w2[f];
    d[i] = acc;
}

// =====================================================================
// Layer-1 PGNN (only the mean path survives in the reference):
//   xp[i,f] = mean_k relu(d[i,k]*p[g[i,k],f] + base[i,f])
// 64 threads per node; coalesced 256B row gathers + prefetch of future rows.
// =====================================================================
__global__ void pgnn_mean_kernel(const float* __restrict__ p, const float* __restrict__ base,
                                 const float* __restrict__ d, const int* __restrict__ garg,
                                 float* __restrict__ xp, int N)
{
    int node = blockIdx.x*4 + (threadIdx.x >> 6);
    int f    = threadIdx.x & 63;
    if (node >= N) return;
    float b   = base[node*64 + f];
    const int*   gi = garg + node*64;
    const float* di = d    + node*64;
    float acc = 0.0f;
    for (int k = 0; k < 64; ++k) {
        if (k + 8 < 64) __builtin_prefetch(&p[(size_t)gi[k+8]*64 + f], 0, 1);
        acc += fmaxf(di[k] * p[(size_t)gi[k]*64 + f] + b, 0.0f);
    }
    xp[node*64 + f] = acc * (1.0f/64.0f);
}

// =====================================================================
// Layer-2 PGNN position head: one wave per (i,k); 32-wide dot with lop2_w.
//   xpos[i,k] = sum_f relu(d2*p2[g,f] + base2[i,f]) * lopw[f]  + lopb
// =====================================================================
__global__ void pgnn_pos_kernel(const float* __restrict__ p2, const float* __restrict__ base2,
                                const float* __restrict__ d2, const int* __restrict__ garg,
                                const float* __restrict__ lopw, const float* __restrict__ lopb,
                                float* __restrict__ xpos, int N)
{
    int wid  = blockIdx.x*(blockDim.x >> 5) + (threadIdx.x >> 5);
    int lane = threadIdx.x & 31;
    int i = wid >> 6, k = wid & 63;
    if (i >= N) return;
    int   g  = garg[i*64 + k];
    float dv = d2[i*64 + k];
    float v = fmaxf(dv * p2[(size_t)g*32 + lane] + base2[i*32 + lane], 0.0f) * lopw[lane];
    #pragma unroll
    for (int off = 16; off > 0; off >>= 1) v += __shfl_xor(v, off, 32);
    if (lane == 0) xpos[i*64 + k] = v + lopb[0];
}

// =====================================================================
// SAGE mean aggregation: scatter-add features + degree counts
// =====================================================================
__global__ void zero_kernel(float* __restrict__ p, int n)
{
    int i = blockIdx.x*blockDim.x + threadIdx.x;
    if (i < n) p[i] = 0.0f;
}

__global__ void sage_scatter_kernel(const float* __restrict__ feat, const int* __restrict__ src,
                                    const int* __restrict__ dst, float* __restrict__ s,
                                    float* __restrict__ cnt, int E, int D)
{
    int idx = blockIdx.x*blockDim.x + threadIdx.x;
    if (idx >= E*D) return;
    int e = idx / D, f = idx % D;
    int so = src[e], de = dst[e];
    atomicAdd(&s[(size_t)de*D + f], feat[(size_t)so*D + f]);
    if (f == 0) atomicAdd(&cnt[de], 1.0f);
}

__global__ void seg_mean_kernel(float* __restrict__ s, const float* __restrict__ cnt, int N, int D)
{
    int idx = blockIdx.x*blockDim.x + threadIdx.x;
    if (idx >= N*D) return;
    s[idx] = s[idx] / fmaxf(cnt[idx / D], 1.0f);
}

// =====================================================================
// Row L2-normalize [N,W] and write into out[:, col0:col0+W] (stride ostride)
// One wave per row, W/32 elements per lane.
// =====================================================================
__global__ void l2norm_out_kernel(const float* __restrict__ x, float* __restrict__ out,
                                  int N, int W, int col0, int ostride)
{
    int row  = blockIdx.x*(blockDim.x >> 5) + (threadIdx.x >> 5);
    int lane = threadIdx.x & 31;
    if (row >= N) return;
    int per = W / 32;
    float vals[2];
    float ss = 0.0f;
    for (int j = 0; j < per; ++j) {
        float v = x[(size_t)row*W + lane*per + j];
        vals[j] = v; ss += v*v;
    }
    #pragma unroll
    for (int off = 16; off > 0; off >>= 1) ss += __shfl_xor(ss, off, 32);
    float inv = 1.0f / fmaxf(sqrtf(ss), 1e-12f);
    for (int j = 0; j < per; ++j)
        out[(size_t)row*ostride + col0 + lane*per + j] = vals[j]*inv;
}

// =====================================================================
extern "C" void kernel_launch(void* const* d_in, const int* in_sizes, int n_in,
                              void* d_out, int out_size, void* d_ws, size_t ws_size,
                              hipStream_t stream)
{
    const float* x        = (const float*)d_in[0];
    const float* dmax     = (const float*)d_in[1];
    const int*   eidx     = (const int*)  d_in[2];
    const int*   garg     = (const int*)  d_in[3];
    const float* w_first  = (const float*)d_in[4];
    const float* b_first  = (const float*)d_in[5];
    const float* w_second = (const float*)d_in[6];
    const float* b_second = (const float*)d_in[7];
    const float* dc1_w1   = (const float*)d_in[8];
    const float* dc1_b1   = (const float*)d_in[9];
    const float* dc1_w2   = (const float*)d_in[10];
    const float* dc1_b2   = (const float*)d_in[11];
    const float* lh1_w    = (const float*)d_in[12];
    const float* lh1_b    = (const float*)d_in[13];
    // d_in[14],d_in[15] = lop1_{w,b}: layer-1 out_pos is discarded by reference
    const float* s1_wl    = (const float*)d_in[16];
    const float* s1_bl    = (const float*)d_in[17];
    const float* s1_wr    = (const float*)d_in[18];
    const float* dc2_w1   = (const float*)d_in[19];
    const float* dc2_b1   = (const float*)d_in[20];
    const float* dc2_w2   = (const float*)d_in[21];
    const float* dc2_b2   = (const float*)d_in[22];
    const float* lh2_w    = (const float*)d_in[23];
    const float* lh2_b    = (const float*)d_in[24];
    const float* lop2_w   = (const float*)d_in[25];
    const float* lop2_b   = (const float*)d_in[26];
    const float* s2_wl    = (const float*)d_in[27];
    const float* s2_bl    = (const float*)d_in[28];
    const float* s2_wr    = (const float*)d_in[29];
    float* out = (float*)d_out;

    const int N = in_sizes[0] / 128;   // 20000 (multiple of 16)
    const int K = 64;
    const int E = in_sizes[2] / 2;     // 320000
    const long long NV = N;

    // Workspace layout (floats). t[N,128] aliases the p1+base1 region
    // (t is fully consumed before p1/base1 are produced).
    float* ws    = (float*)d_ws;
    float* h     = ws;                 // N*64
    float* p1    = h     + NV*64;      // N*64
    float* base1 = p1    + NV*64;      // N*64
    float* t     = p1;                 // N*128 alias
    float* d1    = base1 + NV*64;      // N*64
    float* d2    = d1    + NV*64;      // N*64
    float* xp    = d2    + NV*64;      // N*64
    float* s     = xp    + NV*64;      // N*64 scatter accumulator
    float* cnt   = s     + NV*64;      // N
    float* xs    = cnt   + NV;         // N*64 (x_sage)
    float* p2    = xs    + NV*64;      // N*32
    float* base2 = p2    + NV*32;      // N*32
    float* xpos  = base2 + NV*32;      // N*64
    float* sg2   = xpos  + NV*64;      // N*32

    const int gG = N / 16;             // full tiles only (N % 16 == 0)
    const int* esrc = eidx;
    const int* edst = eidx + E;

    // feature_pre: h = (x@W1 + b1)@W2 + b2
    gemm_wmma_kernel<128,128,false><<<gG,256,0,stream>>>(x, w_first,  nullptr, nullptr, b_first,  t);
    gemm_wmma_kernel<128, 64,false><<<gG,128,0,stream>>>(t, w_second, nullptr, nullptr, b_second, h);

    // layer-1 PGNN (refactored): p1 = h@Wtop, base1 = h@Wbot + b
    gemm_wmma_kernel<64,64,false><<<gG,128,0,stream>>>(h, lh1_w,         nullptr, nullptr, nullptr, p1);
    gemm_wmma_kernel<64,64,false><<<gG,128,0,stream>>>(h, lh1_w + 64*64, nullptr, nullptr, lh1_b,   base1);
    dist_nl_kernel<<<(N*K+255)/256,256,0,stream>>>(dmax, dc1_w1, dc1_b1, dc1_w2, dc1_b2, d1, N*K, 64);
    dist_nl_kernel<<<(N*K+255)/256,256,0,stream>>>(dmax, dc2_w1, dc2_b1, dc2_w2, dc2_b2, d2, N*K, 32);
    pgnn_mean_kernel<<<(N+3)/4,256,0,stream>>>(p1, base1, d1, garg, xp, N);

    // SAGE-1: mean aggregate h, then xs = mean@wl + h@wr + bl (dual WMMA GEMM)
    zero_kernel<<<(N*65+255)/256,256,0,stream>>>(s, N*65);
    sage_scatter_kernel<<<(E*64+255)/256,256,0,stream>>>(h, esrc, edst, s, cnt, E, 64);
    seg_mean_kernel<<<(N*64+255)/256,256,0,stream>>>(s, cnt, N, 64);
    gemm_wmma_kernel<64,64,true><<<gG,128,0,stream>>>(s, s1_wl, h, s1_wr, s1_bl, xs);

    // layer-2 PGNN position head on xp
    gemm_wmma_kernel<64,32,false><<<gG,64,0,stream>>>(xp, lh2_w,         nullptr, nullptr, nullptr, p2);
    gemm_wmma_kernel<64,32,false><<<gG,64,0,stream>>>(xp, lh2_w + 64*32, nullptr, nullptr, lh2_b,   base2);
    pgnn_pos_kernel<<<(N*K+7)/8,256,0,stream>>>(p2, base2, d2, garg, lop2_w, lop2_b, xpos, N);
    l2norm_out_kernel<<<(N+7)/8,256,0,stream>>>(xpos, out, N, 64, 0, 96);

    // SAGE-2: mean aggregate xs, then sg2 = mean@wl2 + xs@wr2 + bl2, L2-norm
    zero_kernel<<<(N*65+255)/256,256,0,stream>>>(s, N*65);
    sage_scatter_kernel<<<(E*64+255)/256,256,0,stream>>>(xs, esrc, edst, s, cnt, E, 64);
    seg_mean_kernel<<<(N*64+255)/256,256,0,stream>>>(s, cnt, N, 64);
    gemm_wmma_kernel<64,32,true><<<gG,64,0,stream>>>(s, s2_wl, xs, s2_wr, s2_bl, sg2);
    l2norm_out_kernel<<<(N+7)/8,256,0,stream>>>(sg2, out, N, 32, 64, 96);

    (void)n_in; (void)out_size; (void)ws_size;
}
